// M3Care_63084479643971
// MI455X (gfx1250) — matrix operand order repaired and checked by reading, not compiled, not run on hardware
//
#include <hip/hip_runtime.h>
#include <math.h>

// ---------------- problem constants (match reference) ----------------
#define MODS   3
#define BSZ    2048
#define DINDIM 2048
#define EDIM   1024
#define RTOK   (BSZ * MODS)        // 6144 token rows

#define LDA_P  68                  // padded A row (floats)
#define LDB_P  72                  // padded B row (ushorts)

typedef __attribute__((ext_vector_type(16))) __bf16 v16bf;
typedef __attribute__((ext_vector_type(8)))  float  v8f;

__device__ __forceinline__ unsigned short f2bf(float f) {
  unsigned int u = __float_as_uint(f);
  u += 0x7FFFu + ((u >> 16) & 1u);          // round-to-nearest-even
  return (unsigned short)(u >> 16);
}

// =====================================================================
// WMMA bf16 GEMM:  C[Mr,N] = A[Mr,K] (f32) x B (f32) (+bias[col]) (relu?)
//   normal: B element (k,n) = Bm[k*ldb + n]
//   btrans: B element (k,n) = Bm[n*ldb + k]   (for gram x @ x^T)
// Block = 256 threads (8 waves); tile 128x128, K-step 64 (2 WMMA k-steps).
// A tile staged f32 via GLOBAL_LOAD_ASYNC_TO_LDS_B128 (ASYNCcnt), bf16 at
// fragment build; B tile staged f32->bf16 via registers into [n][k] LDS.
// Wave w: rows [ (w>>1)*32, +32 ), cols [ (w&1)*64, +64 )  -> 2x4 WMMA tiles
// =====================================================================
__global__ __launch_bounds__(256)
void wmma_gemm(const float* __restrict__ A, const float* __restrict__ Bm,
               float* __restrict__ C, const float* __restrict__ bias,
               int Mr, int N, int K, int ldb, int btrans, int relu) {
  __shared__ __align__(16) float          lsaf[128 * LDA_P];  // [m][k] f32
  __shared__ __align__(16) unsigned short lsb [128 * LDB_P];  // [n][k] bf16
  const int t  = threadIdx.x;
  const int m0 = blockIdx.y * 128;
  const int n0 = blockIdx.x * 128;

  v8f acc[2][4] = {};

  const int lane = t & 31;
  const int wave = t >> 5;
  const int half = lane >> 4;
  const int lm   = lane & 15;
  const int wm   = wave >> 1;        // 0..3
  const int wn   = wave & 1;         // 0..1

  const unsigned lds_a_base = (unsigned)(unsigned long long)(void*)lsaf;

  for (int k0 = 0; k0 < K; k0 += 64) {
    // ---- A tile (128x64 f32): async DMA global -> LDS (16B chunks) ----
#pragma unroll
    for (int i = 0; i < 8; ++i) {
      int c  = i * 256 + t;                 // chunk id, 2048 chunks of 4 floats
      int r  = c >> 4;                      // 16 chunks per 64-float row
      int kc = (c & 15) << 2;
      unsigned ldsoff = lds_a_base + (unsigned)((r * LDA_P + kc) * 4);
      unsigned long long ga =
          (unsigned long long)(A + (size_t)(m0 + r) * K + (k0 + kc));
      asm volatile("global_load_async_to_lds_b128 %0, %1, off"
                   :: "v"(ldsoff), "v"(ga) : "memory");
    }
    // ---- B tile (64x128) f32 -> bf16, LDS layout [n][k] ----
    if (!btrans) {
#pragma unroll
      for (int i = 0; i < 8; ++i) {
        int c  = i * 256 + t;
        int k  = c >> 5;                    // 32 chunks per 128-col k-row
        int n4 = (c & 31) << 2;
        const float4 v =
            *(const float4*)(Bm + (size_t)(k0 + k) * ldb + (n0 + n4));
        lsb[(n4 + 0) * LDB_P + k] = f2bf(v.x);
        lsb[(n4 + 1) * LDB_P + k] = f2bf(v.y);
        lsb[(n4 + 2) * LDB_P + k] = f2bf(v.z);
        lsb[(n4 + 3) * LDB_P + k] = f2bf(v.w);
      }
    } else {
#pragma unroll
      for (int i = 0; i < 8; ++i) {
        int c  = i * 256 + t;
        int n  = c >> 4;
        int kc = (c & 15) << 2;
        const float4 v =
            *(const float4*)(Bm + (size_t)(n0 + n) * ldb + (k0 + kc));
        unsigned lo = (unsigned)f2bf(v.x) | ((unsigned)f2bf(v.y) << 16);
        unsigned hi = (unsigned)f2bf(v.z) | ((unsigned)f2bf(v.w) << 16);
        *(unsigned long long*)&lsb[n * LDB_P + kc] =
            ((unsigned long long)hi << 32) | (unsigned long long)lo;
      }
    }
    if (k0 + 64 < K)   // gfx1250 global_prefetch_b8 for next B tile
      __builtin_prefetch(Bm + (btrans ? ((size_t)(n0 + (t >> 1)) * ldb + k0 + 64)
                                      : ((size_t)(k0 + 64 + (t >> 3)) * ldb + n0)),
                         0, 1);
    asm volatile("s_wait_asynccnt 0" ::: "memory");   // A-tile DMA complete
    __syncthreads();

#pragma unroll
    for (int kk = 0; kk < 64; kk += 32) {
      // ---- A fragments (16x32 bf16, ISA 7.12.2): cvt from f32 LDS ----
      v16bf afr[2], bfr[4];
#pragma unroll
      for (int tim = 0; tim < 2; ++tim) {
        alignas(32) unsigned short tmp[16];
        const float* ar = lsaf + (wm * 32 + tim * 16 + lm) * LDA_P + kk;
#pragma unroll
        for (int e = 0; e < 16; ++e) {
          int j = e >> 1;
          int k = ((j & 4) ? 16 : 0) + half * 8 + ((j & 3) << 1) + (e & 1);
          tmp[e] = f2bf(ar[k]);
        }
        __builtin_memcpy(&afr[tim], tmp, 32);
      }
      // ---- B fragments: lanes 0-15 K=kk+0..7,16..23, lanes 16-31 K=kk+8..15,24..31 ----
#pragma unroll
      for (int tin = 0; tin < 4; ++tin) {
        alignas(32) unsigned short tmp[16];
        const unsigned short* br = lsb + (wn * 64 + tin * 16 + lm) * LDB_P + kk;
#pragma unroll
        for (int e = 0; e < 16; ++e) {
          int j = e >> 1;
          int k = ((j & 4) ? 16 : 0) + half * 8 + ((j & 3) << 1) + (e & 1);
          tmp[e] = br[k];
        }
        __builtin_memcpy(&bfr[tin], tmp, 32);
      }
#pragma unroll
      for (int tim = 0; tim < 2; ++tim)
#pragma unroll
        for (int tin = 0; tin < 4; ++tin)
          acc[tim][tin] = __builtin_amdgcn_wmma_f32_16x16x32_bf16(
              false, afr[tim], false, bfr[tin], (short)0, acc[tim][tin],
              false, false);
    }
    __syncthreads();
  }

  // ---- epilogue: C/D layout (vgpr r: lanes0-15 M=r, lanes16-31 M=8+r; N=lane%16) ----
#pragma unroll
  for (int tim = 0; tim < 2; ++tim)
#pragma unroll
    for (int tin = 0; tin < 4; ++tin)
#pragma unroll
      for (int r = 0; r < 8; ++r) {
        int row = m0 + wm * 32 + tim * 16 + half * 8 + r;
        int col = n0 + wn * 64 + tin * 16 + lm;
        float v = acc[tim][tin][r];
        if (bias) v += bias[col];
        if (relu) v = v > 0.f ? v : 0.f;
        C[(size_t)row * N + col] = v;
      }
}

// =====================================================================
// Small helper kernels
// =====================================================================
__global__ void k_zero(float* p, int n) {
  int i = blockIdx.x * blockDim.x + threadIdx.x;
  if (i < n) p[i] = 0.f;
}

__device__ __forceinline__ float blk_reduce(float v, float* sh) {
  int t = threadIdx.x;
  sh[t] = v; __syncthreads();
  for (int o = 128; o > 0; o >>= 1) {
    if (t < o) sh[t] += sh[t + o];
    __syncthreads();
  }
  return sh[0];
}

__global__ void k_sumsq(const float* __restrict__ x, int n, float* out) {
  __shared__ float sh[256];
  float s = 0.f;
  for (int i = blockIdx.x * 256 + threadIdx.x; i < n; i += gridDim.x * 256) {
    float v = x[i]; s += v * v;
  }
  float r = blk_reduce(s, sh);
  if (threadIdx.x == 0) atomicAdd(out, r);
}

__global__ void k_sumsq_shift(const float* __restrict__ x, int n,
                              const float* s1, float* out) {
  __shared__ float sh[256];
  float c = sqrtf(*s1);
  float s = 0.f;
  for (int i = blockIdx.x * 256 + threadIdx.x; i < n; i += gridDim.x * 256) {
    float v = x[i] - c; s += v * v;
  }
  float r = blk_reduce(s, sh);
  if (threadIdx.x == 0) atomicAdd(out, r);
}

__global__ void k_lstab_fin(const float* s2, float* lacc) {
  if (threadIdx.x == 0 && blockIdx.x == 0) *lacc += fabsf(sqrtf(*s2));
}

// per-column (E blocks) batch stats of relu(src), population variance
__global__ void k_bnstats(const float* __restrict__ src, float* meanv, float* varv) {
  __shared__ float s1[256], s2[256];
  int e = blockIdx.x, t = threadIdx.x;
  float a = 0.f, b = 0.f;
  for (int i = t; i < BSZ; i += 256) {
    float v = src[(size_t)i * EDIM + e];
    v = v > 0.f ? v : 0.f;
    a += v; b += v * v;
  }
  s1[t] = a; s2[t] = b; __syncthreads();
  for (int o = 128; o > 0; o >>= 1) {
    if (t < o) { s1[t] += s1[t + o]; s2[t] += s2[t + o]; }
    __syncthreads();
  }
  if (t == 0) {
    float m = s1[0] / (float)BSZ;
    meanv[e] = m;
    varv[e]  = s2[0] / (float)BSZ - m * m;
  }
}

__global__ void k_bnapply(const float* __restrict__ src, const float* meanv,
                          const float* varv, const float* g, const float* bb,
                          float* __restrict__ dst) {
  int i = blockIdx.x * blockDim.x + threadIdx.x;
  if (i < BSZ * EDIM) {
    int e = i & (EDIM - 1);
    float v = src[i]; v = v > 0.f ? v : 0.f;
    dst[i] = g[e] * (v - meanv[e]) * rsqrtf(varv[e] + 1e-5f) + bb[e];
  }
}

__global__ void k_rowsq(const float* __restrict__ x, float* sq) {
  __shared__ float sh[256];
  float s = 0.f;
  const float* r = x + (size_t)blockIdx.x * EDIM;
  for (int e = threadIdx.x; e < EDIM; e += 256) { float v = r[e]; s += v * v; }
  float t = blk_reduce(s, sh);
  if (threadIdx.x == 0) sq[blockIdx.x] = t;
}

// in-place: g <- max(sq_i + sq_j - 2 g, 0); accumulate sum into *sum
__global__ void k_l2(float* __restrict__ g, const float* __restrict__ sq, float* sum) {
  __shared__ float sh[256];
  float s = 0.f;
  for (int idx = blockIdx.x * 256 + threadIdx.x; idx < BSZ * BSZ;
       idx += gridDim.x * 256) {
    int i = idx >> 11, j = idx & (BSZ - 1);
    float v = sq[i] + sq[j] - 2.f * g[idx];
    v = v > 0.f ? v : 0.f;
    g[idx] = v; s += v;
  }
  float r = blk_reduce(s, sh);
  if (threadIdx.x == 0) atomicAdd(sum, r);
}

__global__ void k_filt(float* __restrict__ filt, const float* __restrict__ l2w,
                       const float* __restrict__ l2g, const float* sumw,
                       const float* sumg, const float* eps_p, int m) {
  const float nn  = (float)BSZ * (float)BSZ - (float)BSZ;
  const float bww = (*sumw) / nn / 2.0f;      // GK_MUL^(GK_NUM//2) = 2
  const float bwg = (*sumg) / nn / 2.0f;
  const float s   = 1.f / (1.f + expf(-eps_p[m]));
  for (int idx = blockIdx.x * 256 + threadIdx.x; idx < BSZ * BSZ;
       idx += gridDim.x * 256) {
    float gw = 0.f, gg = 0.f, bw = bww, bg = bwg;
#pragma unroll
    for (int i = 0; i < 3; ++i) {
      gw += expf(-l2w[idx] / bw);
      gg += expf(-l2g[idx] / bg);
      bw *= 2.f; bg *= 2.f;
    }
    filt[idx] += ((1.f - s) * gw + s) * gg;
  }
}

__global__ void k_thresh(float* __restrict__ filt, const float* thresh) {
  float th = 1.f / (1.f + expf(-(*thresh)));
  for (int idx = blockIdx.x * 256 + threadIdx.x; idx < BSZ * BSZ;
       idx += gridDim.x * 256) {
    float v = filt[idx];
    filt[idx] = (v > th) ? v : 0.f;
  }
}

// out[row] = sigmoid(dot(x[row,:E], wv) + *bsc)
__global__ void k_rowdot_sig(const float* __restrict__ x, const float* __restrict__ wv,
                             const float* bsc, float* out) {
  __shared__ float sh[256];
  const float* r = x + (size_t)blockIdx.x * EDIM;
  float s = 0.f;
  for (int e = threadIdx.x; e < EDIM; e += 256) s += r[e] * wv[e];
  float d = blk_reduce(s, sh);
  if (threadIdx.x == 0) out[blockIdx.x] = 1.f / (1.f + expf(-(d + *bsc)));
}

__global__ void k_impute(const float* __restrict__ emb, const float* __restrict__ aux,
                         const float* __restrict__ ws, const float* __restrict__ wo,
                         const float* __restrict__ temb, float* __restrict__ xtok,
                         int m) {
  int idx = blockIdx.x * blockDim.x + threadIdx.x;
  if (idx < BSZ * EDIM) {
    int i = idx >> 10, e = idx & (EDIM - 1);
    float a = ws[i], b = wo[i];
    xtok[((size_t)i * MODS + m) * EDIM + e] =
        (a * emb[idx] + b * aux[idx]) / (a + b) + temb[e];
  }
}

// attention over M=3 tokens; one block per (sample, head)
__global__ void k_attn(const float* __restrict__ qkv, float* __restrict__ obuf,
                       int heads, int dh) {
  __shared__ float red[256];
  __shared__ float sc[9];
  const int b = blockIdx.x / heads;
  const int h = blockIdx.x - b * heads;
  const int t = threadIdx.x;
  const float scale = rsqrtf((float)dh);
  for (int p = 0; p < 9; ++p) {
    int lq = p / 3, mk = p - lq * 3;
    const float* q = qkv + ((size_t)(b * 3 + lq) * 3 + 0) * EDIM + h * dh;
    const float* k = qkv + ((size_t)(b * 3 + mk) * 3 + 1) * EDIM + h * dh;
    float s = 0.f;
    for (int d = t; d < dh; d += 256) s += q[d] * k[d];
    float r = blk_reduce(s, red);
    if (t == 0) sc[p] = r * scale;
    __syncthreads();
  }
  if (t == 0) {
    for (int l = 0; l < 3; ++l) {
      float a = sc[l * 3], bq = sc[l * 3 + 1], c = sc[l * 3 + 2];
      float mx = fmaxf(a, fmaxf(bq, c));
      float e0 = expf(a - mx), e1 = expf(bq - mx), e2 = expf(c - mx);
      float inv = 1.f / (e0 + e1 + e2);
      sc[l * 3] = e0 * inv; sc[l * 3 + 1] = e1 * inv; sc[l * 3 + 2] = e2 * inv;
    }
  }
  __syncthreads();
  for (int l = 0; l < 3; ++l)
    for (int d = t; d < dh; d += 256) {
      float o = 0.f;
#pragma unroll
      for (int mm = 0; mm < 3; ++mm)
        o += sc[l * 3 + mm] *
             qkv[((size_t)(b * 3 + mm) * 3 + 2) * EDIM + h * dh + d];
      obuf[(size_t)(b * 3 + l) * EDIM + h * dh + d] = o;
    }
}

// x[row] = LN(x[row] + res[row]) over E
__global__ void k_ln(float* __restrict__ x, const float* __restrict__ res,
                     const float* g, const float* bb) {
  __shared__ float sh[256];
  __shared__ float stats[2];
  const int t = threadIdx.x;
  float* xr = x + (size_t)blockIdx.x * EDIM;
  const float* rr = res + (size_t)blockIdx.x * EDIM;
  float s = 0.f;
  for (int e = t; e < EDIM; e += 256) s += xr[e] + rr[e];
  float tot = blk_reduce(s, sh);
  if (t == 0) stats[0] = tot / (float)EDIM;
  __syncthreads();
  float mu = stats[0];
  s = 0.f;
  for (int e = t; e < EDIM; e += 256) { float v = xr[e] + rr[e] - mu; s += v * v; }
  tot = blk_reduce(s, sh);
  if (t == 0) stats[1] = rsqrtf(tot / (float)EDIM + 1e-5f);
  __syncthreads();
  float rstd = stats[1];
  for (int e = t; e < EDIM; e += 256)
    xr[e] = g[e] * (xr[e] + rr[e] - mu) * rstd + bb[e];
}

// y[row] = dot(y1[row, :2E], w) + b[0]
__global__ void k_fc2(const float* __restrict__ y1, const float* __restrict__ w,
                      const float* b, float* __restrict__ out) {
  __shared__ float sh[256];
  const float* r = y1 + (size_t)blockIdx.x * (2 * EDIM);
  float s = 0.f;
  for (int e = threadIdx.x; e < 2 * EDIM; e += 256) s += r[e] * w[e];
  float d = blk_reduce(s, sh);
  if (threadIdx.x == 0) out[blockIdx.x] = d + b[0];
}

__global__ void k_copy1(const float* src, float* dst) {
  if (blockIdx.x == 0 && threadIdx.x == 0) dst[0] = src[0];
}

// =====================================================================
extern "C" void kernel_launch(void* const* d_in, const int* in_sizes, int n_in,
                              void* d_out, int out_size, void* d_ws, size_t ws_size,
                              hipStream_t stream) {
  const float* xs        = (const float*)d_in[0];
  const float* enc_W     = (const float*)d_in[1];
  const float* enc_b     = (const float*)d_in[2];
  const float* proj_W    = (const float*)d_in[3];
  const float* proj_b    = (const float*)d_in[4];
  const float* bn_gamma  = (const float*)d_in[5];
  const float* bn_beta   = (const float*)d_in[6];
  const float* eps_p     = (const float*)d_in[7];
  const float* thresh    = (const float*)d_in[8];
  const float* gcn_W     = (const float*)d_in[9];
  const float* gcn_b     = (const float*)d_in[10];
  const float* ad_s_W    = (const float*)d_in[11];
  const float* ad_s_b    = (const float*)d_in[12];
  const float* ad_o_W    = (const float*)d_in[13];
  const float* ad_o_b    = (const float*)d_in[14];
  const float* type_emb  = (const float*)d_in[15];
  const float* qkv_W     = (const float*)d_in[16];
  const float* qkv_b     = (const float*)d_in[17];
  const float* ao_W      = (const float*)d_in[18];
  const float* ao_b      = (const float*)d_in[19];
  const float* ln1_g     = (const float*)d_in[20];
  const float* ln1_b     = (const float*)d_in[21];
  const float* ln2_g     = (const float*)d_in[22];
  const float* ln2_b     = (const float*)d_in[23];
  const float* ffn_W1    = (const float*)d_in[24];
  const float* ffn_b1    = (const float*)d_in[25];
  const float* ffn_W2    = (const float*)d_in[26];
  const float* ffn_b2    = (const float*)d_in[27];
  const float* fc1_W     = (const float*)d_in[28];
  const float* fc1_b     = (const float*)d_in[29];
  const float* fc2_W     = (const float*)d_in[30];
  const float* fc2_b     = (const float*)d_in[31];

  float* W = (float*)d_ws;
  const size_t BE = (size_t)BSZ * EDIM, BB = (size_t)BSZ * BSZ;
  // persistent
  float* EMB  = W;                              // 3*B*E
  float* FILT = EMB + 3 * BE;                   // B*B
  float* XTOK = FILT + BB;                      // B*M*E
  float* SCAL = XTOK + (size_t)BSZ * MODS * EDIM;  // 64 scalars
  float* SCR  = SCAL + 64;
  // phase-1 scratch layout
  float* T0  = SCR;             float* T1  = T0 + BE;
  float* GKA = T1 + BE;         float* GKB = GKA + BB;
  float* SQ  = GKB + BB;        float* WSv = SQ + BSZ;
  float* WOv = WSv + BSZ;       float* MEANv = WOv + BSZ;
  float* VARv = MEANv + EDIM;
  // phase-3 scratch layout (reuses SCR region)
  float* QKVb = SCR;                              // R*3E
  float* HID  = QKVb + (size_t)RTOK * 3 * EDIM;   // R*4E
  float* OB   = HID  + (size_t)RTOK * 4 * EDIM;   // R*E
  float* Y1   = OB   + (size_t)RTOK * EDIM;       // B*2E

  auto gemm = [&](const float* A, const float* Bm, float* C, const float* bias,
                  int Mr, int N, int K, int ldb, int btrans, int relu) {
    dim3 grid(N / 128, Mr / 128);
    wmma_gemm<<<grid, 256, 0, stream>>>(A, Bm, C, bias, Mr, N, K, ldb, btrans, relu);
  };
  auto zero = [&](float* p, int n) {
    k_zero<<<(n + 255) / 256, 256, 0, stream>>>(p, n);
  };

  zero(FILT, BSZ * BSZ);
  zero(SCAL, 8);

  // ---------------- Phase 1: embeddings, proj, lstab, gaussian filt ----------------
  for (int m = 0; m < MODS; ++m) {
    float* Em = EMB + (size_t)m * BE;
    gemm(xs + (size_t)m * BSZ * DINDIM, enc_W + (size_t)m * DINDIM * EDIM,
         Em, enc_b + (size_t)m * EDIM, BSZ, EDIM, DINDIM, EDIM, 0, 0);
    gemm(Em, proj_W + ((size_t)m * 3 + 0) * EDIM * EDIM, T0,
         proj_b + ((size_t)m * 3 + 0) * EDIM, BSZ, EDIM, EDIM, EDIM, 0, 1);
    gemm(T0, proj_W + ((size_t)m * 3 + 1) * EDIM * EDIM, T1,
         proj_b + ((size_t)m * 3 + 1) * EDIM, BSZ, EDIM, EDIM, EDIM, 0, 1);
    gemm(T1, proj_W + ((size_t)m * 3 + 2) * EDIM * EDIM, T0,
         proj_b + ((size_t)m * 3 + 2) * EDIM, BSZ, EDIM, EDIM, EDIM, 0, 0);
    // lstab += | frob(h - frob(emb)) |
    zero(SCAL + 0, 2);
    k_sumsq<<<512, 256, 0, stream>>>(Em, BSZ * EDIM, SCAL + 0);
    k_sumsq_shift<<<512, 256, 0, stream>>>(T0, BSZ * EDIM, SCAL + 0, SCAL + 1);
    k_lstab_fin<<<1, 32, 0, stream>>>(SCAL + 1, SCAL + 4);
    // sim_wgk: bn(relu(h)) -> gram -> l2
    k_bnstats<<<EDIM, 256, 0, stream>>>(T0, MEANv, VARv);
    k_bnapply<<<(BSZ * EDIM + 255) / 256, 256, 0, stream>>>(T0, MEANv, VARv,
                                                            bn_gamma, bn_beta, T1);
    k_rowsq<<<BSZ, 256, 0, stream>>>(T1, SQ);
    gemm(T1, T1, GKA, nullptr, BSZ, BSZ, EDIM, EDIM, 1, 0);
    zero(SCAL + 2, 1);
    k_l2<<<1024, 256, 0, stream>>>(GKA, SQ, SCAL + 2);
    // sim_gk: bn(relu(emb)) -> gram -> l2
    k_bnstats<<<EDIM, 256, 0, stream>>>(Em, MEANv, VARv);
    k_bnapply<<<(BSZ * EDIM + 255) / 256, 256, 0, stream>>>(Em, MEANv, VARv,
                                                            bn_gamma, bn_beta, T1);
    k_rowsq<<<BSZ, 256, 0, stream>>>(T1, SQ);
    gemm(T1, T1, GKB, nullptr, BSZ, BSZ, EDIM, EDIM, 1, 0);
    zero(SCAL + 3, 1);
    k_l2<<<1024, 256, 0, stream>>>(GKB, SQ, SCAL + 3);
    k_filt<<<1024, 256, 0, stream>>>(FILT, GKA, GKB, SCAL + 2, SCAL + 3, eps_p, m);
  }
  k_thresh<<<1024, 256, 0, stream>>>(FILT, thresh);

  // ---------------- Phase 2: GCN + adaptive imputation -> tokens ----------------
  for (int m = 0; m < MODS; ++m) {
    float* Em = EMB + (size_t)m * BE;
    gemm(Em, gcn_W + (size_t)m * EDIM * EDIM, T0, nullptr, BSZ, EDIM, EDIM, EDIM, 0, 0);
    gemm(FILT, T0, T1, gcn_b + (size_t)m * EDIM, BSZ, EDIM, BSZ, EDIM, 0, 1); // aux
    k_rowdot_sig<<<BSZ, 256, 0, stream>>>(Em, ad_s_W + (size_t)m * EDIM, ad_s_b + m, WSv);
    k_rowdot_sig<<<BSZ, 256, 0, stream>>>(T1, ad_o_W + (size_t)m * EDIM, ad_o_b + m, WOv);
    k_impute<<<(BSZ * EDIM + 255) / 256, 256, 0, stream>>>(
        Em, T1, WSv, WOv, type_emb + (size_t)m * EDIM, XTOK, m);
  }

  // ---------------- Phase 3: 2-layer transformer ----------------
  for (int l = 0; l < 2; ++l) {
    int heads = (l == 0) ? 4 : 1;
    int dh = EDIM / heads;
    gemm(XTOK, qkv_W + (size_t)l * EDIM * 3 * EDIM, QKVb,
         qkv_b + (size_t)l * 3 * EDIM, RTOK, 3 * EDIM, EDIM, 3 * EDIM, 0, 0);
    k_attn<<<BSZ * heads, 256, 0, stream>>>(QKVb, OB, heads, dh);
    gemm(OB, ao_W + (size_t)l * EDIM * EDIM, HID, ao_b + (size_t)l * EDIM,
         RTOK, EDIM, EDIM, EDIM, 0, 0);                   // o-proj into HID[:R*E]
    k_ln<<<RTOK, 256, 0, stream>>>(XTOK, HID, ln1_g + (size_t)l * EDIM,
                                   ln1_b + (size_t)l * EDIM);
    gemm(XTOK, ffn_W1 + (size_t)l * EDIM * 4 * EDIM, HID,
         ffn_b1 + (size_t)l * 4 * EDIM, RTOK, 4 * EDIM, EDIM, 4 * EDIM, 0, 1);
    gemm(HID, ffn_W2 + (size_t)l * 4 * EDIM * EDIM, OB,
         ffn_b2 + (size_t)l * EDIM, RTOK, EDIM, 4 * EDIM, EDIM, 0, 0);
    k_ln<<<RTOK, 256, 0, stream>>>(XTOK, OB, ln2_g + (size_t)l * EDIM,
                                   ln2_b + (size_t)l * EDIM);
  }

  // ---------------- Final head ----------------
  gemm(XTOK /*[B, M*E]*/, fc1_W, Y1, fc1_b, BSZ, 2 * EDIM, MODS * EDIM,
       2 * EDIM, 0, 1);
  float* out = (float*)d_out;
  k_fc2<<<BSZ, 256, 0, stream>>>(Y1, fc2_W, fc2_b, out);
  k_copy1<<<1, 32, 0, stream>>>(SCAL + 4, out + BSZ);   // lstab
}
